// FullTucker_71648644432007
// MI455X (gfx1250) — matrix operand order repaired and checked by reading, not compiled, not run on hardware
//
#include <hip/hip_runtime.h>

// ---------------------------------------------------------------------------
// FullTucker on MI455X (gfx1250): all GEMMs via v_wmma_f32_16x16x32_bf16,
// Khatri-Rao product folded into the core GEMM with per-lane v_pk_mul_bf16.
// Register double-buffering so fragment loads overlap WMMA issue.
//
// Shapes: D_OUT=D1=D2=512, R0=R1=R2=128, BATCH=8192.
//   UX1 = U1^T X1   (128 x 8192)       [bf16 WMMA, f32 acc]
//   UX2 = U2^T X2   (128 x 8192)
//   core[p,b] = sum_{q,r} G[p,q,r] UX1[q,b] UX2[r,b]
//             = G0[128 x 16384] @ KR   [KR generated on the fly]
//   Z = U0 @ core   (512 x 8192, f32 out)
// ---------------------------------------------------------------------------

typedef __attribute__((ext_vector_type(16))) __bf16 v16bf;
typedef __attribute__((ext_vector_type(8)))  float  v8f;

union Frag {
    v16bf        v;
    unsigned int u[8];
};

__device__ __forceinline__ void load_frag16(Frag& f, const unsigned short* p0,
                                            const unsigned short* p1) {
    const uint4 a = *(const uint4*)p0;   // 8 bf16
    const uint4 b = *(const uint4*)p1;   // 8 bf16
    f.u[0] = a.x; f.u[1] = a.y; f.u[2] = a.z; f.u[3] = a.w;
    f.u[4] = b.x; f.u[5] = b.y; f.u[6] = b.z; f.u[7] = b.w;
}

__device__ __forceinline__ v8f wmma_bf16(const Frag& a, const Frag& b, v8f c) {
    return __builtin_amdgcn_wmma_f32_16x16x32_bf16(
        /*neg_a=*/false, a.v, /*neg_b=*/false, b.v,
        /*c_mod=*/(short)0, c, /*reuse_a=*/false, /*reuse_b=*/false);
}

// Non-volatile: pure value op, scheduler may interleave with WMMA issue.
__device__ __forceinline__ unsigned pk_mul_bf16(unsigned a, unsigned b) {
    unsigned d;
    asm("v_pk_mul_bf16 %0, %1, %2" : "=v"(d) : "v"(a), "v"(b));
    return d;
}

// float -> bf16 (round-to-nearest-even)
__device__ __forceinline__ unsigned short f2bf(float f) {
    unsigned x = __float_as_uint(f);
    return (unsigned short)((x + 0x7fffu + ((x >> 16) & 1u)) >> 16);
}

// ---------------------------------------------------------------------------
// Flat f32 -> bf16 convert (G -> G0 bf16, U0 -> bf16); 4 elems/thread.
// ---------------------------------------------------------------------------
__global__ void cvt_f32_bf16(const float* __restrict__ in,
                             unsigned short* __restrict__ out, int n4) {
    int i = (blockIdx.x * blockDim.x + threadIdx.x);
    if (i < n4) {
        const float4 v = *(const float4*)(in + (size_t)i * 4);
        unsigned short h0 = f2bf(v.x), h1 = f2bf(v.y), h2 = f2bf(v.z), h3 = f2bf(v.w);
        uint2 o;
        o.x = (unsigned)h0 | ((unsigned)h1 << 16);
        o.y = (unsigned)h2 | ((unsigned)h3 << 16);
        *(uint2*)(out + (size_t)i * 4) = o;
    }
}

// ---------------------------------------------------------------------------
// LDS-tiled transpose + convert:  f32 in[rows][cols] -> bf16 out[cols][rows]
// 32x32 tiles, 256 threads, coalesced on both sides.
// ---------------------------------------------------------------------------
__global__ void __launch_bounds__(256)
transpose_cvt(const float* __restrict__ in, unsigned short* __restrict__ out,
              int rows, int cols) {
    __shared__ float tile[32][33];
    const int c0 = blockIdx.x * 32;
    const int r0 = blockIdx.y * 32;
    const int t  = threadIdx.x;
    const int rr  = t >> 3;          // 0..31
    const int cc4 = (t & 7) * 4;     // 0..28

    const float4 v = *(const float4*)(in + (size_t)(r0 + rr) * cols + c0 + cc4);
    tile[rr][cc4 + 0] = v.x;
    tile[rr][cc4 + 1] = v.y;
    tile[rr][cc4 + 2] = v.z;
    tile[rr][cc4 + 3] = v.w;
    __syncthreads();

    // out row = c0+rr, cols r0+cc4 .. +3
    unsigned short h0 = f2bf(tile[cc4 + 0][rr]);
    unsigned short h1 = f2bf(tile[cc4 + 1][rr]);
    unsigned short h2 = f2bf(tile[cc4 + 2][rr]);
    unsigned short h3 = f2bf(tile[cc4 + 3][rr]);
    uint2 o;
    o.x = (unsigned)h0 | ((unsigned)h1 << 16);
    o.y = (unsigned)h2 | ((unsigned)h3 << 16);
    *(uint2*)(out + (size_t)(c0 + rr) * rows + r0 + cc4) = o;
}

// ---------------------------------------------------------------------------
// Store one 16x16 f32 accumulator tile transposed as bf16 [b][m] (b-major).
// D layout: VGPR v -> (M = v + 8*(lane>>4), N = lane&15).
// ---------------------------------------------------------------------------
__device__ __forceinline__ void store_tile_bf16_T(unsigned short* __restrict__ dst,
                                                  int bRow, int mCol, int l16, int lhi,
                                                  const v8f& acc, int ldm) {
    unsigned short h[8];
#pragma unroll
    for (int v = 0; v < 8; ++v) h[v] = f2bf(acc[v]);
    uint4 o;
    o.x = (unsigned)h[0] | ((unsigned)h[1] << 16);
    o.y = (unsigned)h[2] | ((unsigned)h[3] << 16);
    o.z = (unsigned)h[4] | ((unsigned)h[5] << 16);
    o.w = (unsigned)h[6] | ((unsigned)h[7] << 16);
    *(uint4*)(dst + (size_t)(bRow + l16) * ldm + mCol + lhi * 8) = o;
}

// ---------------------------------------------------------------------------
// Projection GEMM: UXt[b][q] = sum_k Ut[q][k] * Xt[b][k]
//   Ut : bf16 [128][512]  (= U^T, row q contiguous)  -> A operand
//   Xt : bf16 [8192][512] (= X^T, row b contiguous)  -> B operand
//   UXt: bf16 [8192][128] (b-major, feeds next stage)
// Block: 256 thr (8 waves), wave w -> q-tile 16w, N-tile = 64 per block.
// Ping-pong double buffering over k0 so loads overlap WMMA.
// ---------------------------------------------------------------------------
__device__ __forceinline__ void proj_load(Frag& a, Frag b[4],
                                          const unsigned short* __restrict__ arow,
                                          const unsigned short* __restrict__ Xt,
                                          int n0, int l16, int lhi, int k0) {
    load_frag16(a, arow + k0, arow + k0 + 8);
#pragma unroll
    for (int s = 0; s < 4; ++s) {
        const unsigned short* bp =
            Xt + (size_t)(n0 + s * 16 + l16) * 512 + k0 + lhi * 16;
        load_frag16(b[s], bp, bp + 8);
    }
}

__global__ void __launch_bounds__(256)
proj_gemm(const unsigned short* __restrict__ Ut,
          const unsigned short* __restrict__ Xt,
          unsigned short* __restrict__ UXt) {
    const int tid = threadIdx.x, wave = tid >> 5, lane = tid & 31;
    const int l16 = lane & 15, lhi = lane >> 4;
    const int m0 = wave * 16;
    const int n0 = blockIdx.x * 64;

    v8f acc[4] = {};
    const unsigned short* arow = Ut + (size_t)(m0 + l16) * 512 + lhi * 16;

    Frag aA, bA[4], aB, bB[4];
    proj_load(aA, bA, arow, Xt, n0, l16, lhi, 0);

#pragma unroll
    for (int k0 = 0; k0 < 512; k0 += 64) {
        proj_load(aB, bB, arow, Xt, n0, l16, lhi, k0 + 32);
#pragma unroll
        for (int s = 0; s < 4; ++s) acc[s] = wmma_bf16(aA, bA[s], acc[s]);
        if (k0 + 64 < 512) proj_load(aA, bA, arow, Xt, n0, l16, lhi, k0 + 64);
#pragma unroll
        for (int s = 0; s < 4; ++s) acc[s] = wmma_bf16(aB, bB[s], acc[s]);
    }
#pragma unroll
    for (int s = 0; s < 4; ++s)
        store_tile_bf16_T(UXt, n0 + s * 16, m0, l16, lhi, acc[s], 128);
}

// ---------------------------------------------------------------------------
// Core GEMM with on-the-fly Khatri-Rao B operand:
//   coret[b][p] = sum_q sum_r G0[p][q*128+r] * (UX1t[b][q] * UX2t[b][r])
// Wave holds all 16 UX2 B-fragments (4 r0-slices x 4 N-subtiles) in VGPRs;
// per q: scale by broadcast UX1[q,b] via v_pk_mul_bf16, stream G0 rows as A.
// A fragments + UX1 scalars ping-pong double-buffered over q.
// ---------------------------------------------------------------------------
__device__ __forceinline__ void core_loadA(Frag a[4], unsigned ux1p[4],
                                           const unsigned short* __restrict__ arow,
                                           const unsigned short* __restrict__ UX1t,
                                           int n0, int l16, int q) {
    const unsigned short* ak = arow + q * 128;
#pragma unroll
    for (int r = 0; r < 4; ++r)
        load_frag16(a[r], ak + r * 32, ak + r * 32 + 8);
#pragma unroll
    for (int s = 0; s < 4; ++s) {
        unsigned short t = UX1t[(size_t)(n0 + s * 16 + l16) * 128 + q];
        ux1p[s] = (unsigned)t * 0x10001u;   // duplicate bf16 into both halves
    }
}

__device__ __forceinline__ void core_step(const Frag a[4], const unsigned ux1p[4],
                                          const Frag bx[4][4], v8f acc[4]) {
#pragma unroll
    for (int r = 0; r < 4; ++r)
#pragma unroll
        for (int s = 0; s < 4; ++s) {
            Frag bp;  // KR fragment = UX2 fragment (*) UX1 broadcast
#pragma unroll
            for (int i = 0; i < 8; ++i)
                bp.u[i] = pk_mul_bf16(bx[r][s].u[i], ux1p[s]);
            acc[s] = wmma_bf16(a[r], bp, acc[s]);
        }
}

__global__ void __launch_bounds__(256)
core_gemm(const unsigned short* __restrict__ G0,    // [128][16384]
          const unsigned short* __restrict__ UX1t,  // [8192][128]
          const unsigned short* __restrict__ UX2t,  // [8192][128]
          unsigned short* __restrict__ coret) {     // [8192][128]
    const int tid = threadIdx.x, wave = tid >> 5, lane = tid & 31;
    const int l16 = lane & 15, lhi = lane >> 4;
    const int m0 = wave * 16;          // R0 tile (M)
    const int n0 = blockIdx.x * 64;    // batch tile (N)

    // Preload UX2 B-fragments: bx[r0/32][sub] covers K(=r) slice r0..r0+31,
    // columns b = n0 + sub*16 .. +15. Stays resident for the whole q loop.
    Frag bx[4][4];
#pragma unroll
    for (int r = 0; r < 4; ++r)
#pragma unroll
        for (int s = 0; s < 4; ++s) {
            const unsigned short* p =
                UX2t + (size_t)(n0 + s * 16 + l16) * 128 + r * 32 + lhi * 16;
            load_frag16(bx[r][s], p, p + 8);
        }

    v8f acc[4] = {};
    const unsigned short* arow = G0 + (size_t)(m0 + l16) * 16384 + lhi * 16;

    Frag aA[4], aB[4];
    unsigned uA[4], uB[4];
    core_loadA(aA, uA, arow, UX1t, n0, l16, 0);

    for (int q = 0; q < 128; q += 2) {
        core_loadA(aB, uB, arow, UX1t, n0, l16, q + 1);
        core_step(aA, uA, bx, acc);
        if (q + 2 < 128) core_loadA(aA, uA, arow, UX1t, n0, l16, q + 2);
        core_step(aB, uB, bx, acc);
    }
#pragma unroll
    for (int s = 0; s < 4; ++s)
        store_tile_bf16_T(coret, n0 + s * 16, m0, l16, lhi, acc[s], 128);
}

// ---------------------------------------------------------------------------
// Output GEMM: Z[m][b] = sum_k U0b[m][k] * coret[b][k]   (f32 output)
// K=128 -> fully unrolled, all fragments loaded before the WMMA block.
// grid.x over batch (64/block), grid.y over D_OUT (128/block).
// ---------------------------------------------------------------------------
__global__ void __launch_bounds__(256)
out_gemm(const unsigned short* __restrict__ U0b,   // [512][128]
         const unsigned short* __restrict__ coret, // [8192][128]
         float* __restrict__ Z) {                  // [512][8192]
    const int tid = threadIdx.x, wave = tid >> 5, lane = tid & 31;
    const int l16 = lane & 15, lhi = lane >> 4;
    const int m0 = blockIdx.y * 128 + wave * 16;
    const int n0 = blockIdx.x * 64;

    const unsigned short* arow = U0b + (size_t)(m0 + l16) * 128 + lhi * 16;

    Frag a[4], b[4][4];
#pragma unroll
    for (int k = 0; k < 4; ++k) {
        load_frag16(a[k], arow + k * 32, arow + k * 32 + 8);
#pragma unroll
        for (int s = 0; s < 4; ++s) {
            const unsigned short* bp =
                coret + (size_t)(n0 + s * 16 + l16) * 128 + k * 32 + lhi * 16;
            load_frag16(b[k][s], bp, bp + 8);
        }
    }

    v8f acc[4] = {};
#pragma unroll
    for (int k = 0; k < 4; ++k)
#pragma unroll
        for (int s = 0; s < 4; ++s)
            acc[s] = wmma_bf16(a[k], b[k][s], acc[s]);

#pragma unroll
    for (int s = 0; s < 4; ++s) {
        const int bcol = n0 + s * 16 + l16;
#pragma unroll
        for (int v = 0; v < 8; ++v)
            Z[(size_t)(m0 + v + lhi * 8) * 8192 + bcol] = acc[s][v];
    }
}

// ---------------------------------------------------------------------------
// Launch
// ---------------------------------------------------------------------------
extern "C" void kernel_launch(void* const* d_in, const int* in_sizes, int n_in,
                              void* d_out, int out_size, void* d_ws, size_t ws_size,
                              hipStream_t stream) {
    const float* X1 = (const float*)d_in[0];  // [512][8192]
    const float* X2 = (const float*)d_in[1];  // [512][8192]
    const float* U0 = (const float*)d_in[2];  // [512][128]
    const float* U1 = (const float*)d_in[3];  // [512][128]
    const float* U2 = (const float*)d_in[4];  // [512][128]
    const float* G  = (const float*)d_in[5];  // [128][128][128]
    float* Z = (float*)d_out;                 // [512][8192]

    // Workspace layout (bf16 buffers), total ~26.4 MB
    char* ws = (char*)d_ws;
    size_t off = 0;
    unsigned short* Gb   = (unsigned short*)(ws + off); off += (size_t)128 * 16384 * 2;
    unsigned short* U0b  = (unsigned short*)(ws + off); off += (size_t)512 * 128 * 2;
    unsigned short* U1t  = (unsigned short*)(ws + off); off += (size_t)128 * 512 * 2;
    unsigned short* U2t  = (unsigned short*)(ws + off); off += (size_t)128 * 512 * 2;
    unsigned short* X1t  = (unsigned short*)(ws + off); off += (size_t)8192 * 512 * 2;
    unsigned short* X2t  = (unsigned short*)(ws + off); off += (size_t)8192 * 512 * 2;
    unsigned short* UX1t = (unsigned short*)(ws + off); off += (size_t)8192 * 128 * 2;
    unsigned short* UX2t = (unsigned short*)(ws + off); off += (size_t)8192 * 128 * 2;
    unsigned short* Ct   = (unsigned short*)(ws + off); off += (size_t)8192 * 128 * 2;
    if (ws_size < off) return;  // workspace too small; deterministic no-op

    // Stage 0: convert G -> G0 (flat layout == mode-0 unfolding), U0 -> bf16
    cvt_f32_bf16<<<(128 * 16384 / 4 + 255) / 256, 256, 0, stream>>>(G, Gb, 128 * 16384 / 4);
    cvt_f32_bf16<<<(512 * 128 / 4 + 255) / 256, 256, 0, stream>>>(U0, U0b, 512 * 128 / 4);

    // Stage 1: transpose+convert U1,U2 (->[128][512]) and X1,X2 (->[8192][512])
    transpose_cvt<<<dim3(128 / 32, 512 / 32), 256, 0, stream>>>(U1, U1t, 512, 128);
    transpose_cvt<<<dim3(128 / 32, 512 / 32), 256, 0, stream>>>(U2, U2t, 512, 128);
    transpose_cvt<<<dim3(8192 / 32, 512 / 32), 256, 0, stream>>>(X1, X1t, 512, 8192);
    transpose_cvt<<<dim3(8192 / 32, 512 / 32), 256, 0, stream>>>(X2, X2t, 512, 8192);

    // Stage 2: projections UXi = Ui^T Xi (stored b-major bf16)
    proj_gemm<<<8192 / 64, 256, 0, stream>>>(U1t, X1t, UX1t);
    proj_gemm<<<8192 / 64, 256, 0, stream>>>(U2t, X2t, UX2t);

    // Stage 3: core = G0 @ khatri_rao(UX1, UX2), KR generated in-register
    core_gemm<<<8192 / 64, 256, 0, stream>>>(Gb, UX1t, UX2t, Ct);

    // Stage 4: Z = U0 @ core (f32 out)
    out_gemm<<<dim3(8192 / 64, 512 / 128), 256, 0, stream>>>(U0b, Ct, Z);
}
